// unit_gcn_48765058679538
// MI455X (gfx1250) — compile-verified, matrix operand order checked by reading, not run on hardware
//
#include <hip/hip_runtime.h>
#include <hip/hip_bf16.h>

// ---------------------------------------------------------------------------
// unit-GCN forward on gfx1250 (MI455X), bf16 WMMA pipeline.
// N=64, C=64, T=256, V=25, D=64.  NT=16384, Q=T*V=6400, M=N*Q=409600.
// ---------------------------------------------------------------------------

#define N_   64
#define C_   64
#define T_   256
#define V_   25
#define D_   64
#define Q_   6400      // T_*V_
#define NT_  16384     // N_*T_
#define M_   409600    // N_*Q_

typedef __attribute__((ext_vector_type(16))) __bf16 v16bf;
typedef __attribute__((ext_vector_type(8)))  __bf16 v8bf;
typedef __attribute__((ext_vector_type(8)))  float  v8f;

__device__ inline __bf16 f2bf(float f) { return (__bf16)f; }  // native v_cvt

__device__ inline v8f wmma_bf16(v16bf a, v16bf b, v8f c) {
    return __builtin_amdgcn_wmma_f32_16x16x32_bf16(
        false, a, false, b, (short)0, c, false, false);
}

// B-fragment (32x16 bf16) from row-major [col][K], row stride 64 (LDS).
__device__ inline v16bf ldb64(const __bf16* row, int k0, int hi) {
    v16bf f;
    const __bf16* p = row + k0 + hi * 8;
#pragma unroll
    for (int e = 0; e < 8; e++) { f[e] = p[e]; f[8 + e] = p[16 + e]; }
    return f;
}

// ---------------------------------------------------------------------------
// Kernel 1: x = bn0(x0 @ lin_w + lin_b)  -> X  [D][NT][V]  (bf16, b128 stores)
//           a = x0 @ wa^T + ba           -> A1 [D][V][NT]  (bf16, LDS-restaged)
//           b = x0 @ wb^T + bb           -> B1 [D][V][NT]  (bf16, LDS-restaged)
// One workgroup = 400 consecutive q (16 t x 25 v, one n) = 25 row-tiles of 16.
// Dynamic LDS: weights (24.5 KB) + scalars (1 KB) + LA/LB (2 x 52 KB).
// ---------------------------------------------------------------------------
__global__ __launch_bounds__(256) void k_gemm3(
    const float* __restrict__ x0, const float* __restrict__ lin_w,
    const float* __restrict__ lin_b,
    const float* __restrict__ wa, const float* __restrict__ ba,
    const float* __restrict__ wb, const float* __restrict__ bb,
    const float* __restrict__ bn0g, const float* __restrict__ bn0b,
    const float* __restrict__ bn0m, const float* __restrict__ bn0v,
    __bf16* __restrict__ X, __bf16* __restrict__ A1, __bf16* __restrict__ B1)
{
    extern __shared__ char smem[];
    float*  xs  = (float*)smem;                 // [64] bn0 scale
    float*  xh  = xs + 64;                      // [64] bn0 shift (lin_b folded)
    float*  baS = xh + 64;                      // [64]
    float*  bbS = baS + 64;                     // [64]
    __bf16* lwT = (__bf16*)(smem + 1024);       // [64][64] lin_w^T  [d][c]
    __bf16* waS = lwT + 4096;                   // [64][64] wa       [d][c]
    __bf16* wbS = waS + 4096;                   // [64][64] wb       [d][c]
    __bf16* LA  = (__bf16*)(smem + 25600);      // [400][65] a tile (padded)
    __bf16* LB  = LA + 400 * 65;                // [400][65] b tile

    int tid = threadIdx.x;
    for (int i = tid; i < 4096; i += 256) {
        int dd = i >> 6, cc = i & 63;
        lwT[i] = f2bf(lin_w[cc * 64 + dd]);
        waS[i] = f2bf(wa[i]);
        wbS[i] = f2bf(wb[i]);
    }
    if (tid < 64) {
        float inv = bn0g[tid] * rsqrtf(bn0v[tid] + 1e-5f);
        xs[tid] = inv;
        xh[tid] = lin_b[tid] * inv + (bn0b[tid] - bn0m[tid] * inv);
        baS[tid] = ba[tid];
        bbS[tid] = bb[tid];
    }
    __syncthreads();

    int wave = tid >> 5, lane = tid & 31, r = lane & 15, hi = lane >> 4;
    int base = blockIdx.x * 400;           // 1024 blocks
    int n    = base / Q_;
    int qb   = base % Q_;                  // multiple of 400 -> t-block aligned
    int tb   = qb / V_;                    // 16-aligned t start
    int ntb  = n * T_ + tb;

    for (int tl = wave; tl < 25; tl += 8) {       // 25 tiles of 16 rows
        int q0 = qb + tl * 16;
        // A fragments: A[m,c] = x0[n,c,q]; coalesced 64B per element index.
        v16bf af[2];
        const float* abase = x0 + n * (C_ * Q_) + q0 + r;
#pragma unroll
        for (int ks = 0; ks < 2; ks++)
#pragma unroll
            for (int e = 0; e < 16; e++) {
                int c = ks * 32 + ((e & 8) << 1) + hi * 8 + (e & 7);
                af[ks][e] = f2bf(abase[c * Q_]);
            }

#pragma unroll
        for (int dt = 0; dt < 4; dt++) {
            int d0 = dt * 16;
            v8f accx = {}, acca = {}, accb = {};
#pragma unroll
            for (int ks = 0; ks < 2; ks++) {
                accx = wmma_bf16(af[ks], ldb64(lwT + (d0 + r) * 64, ks * 32, hi), accx);
                acca = wmma_bf16(af[ks], ldb64(waS + (d0 + r) * 64, ks * 32, hi), acca);
                accb = wmma_bf16(af[ks], ldb64(wbS + (d0 + r) * 64, ks * 32, hi), accb);
            }
            int   d  = d0 + r;                    // D-frag: col N = lane&15
            float sx = xs[d], hx = xh[d], av = baS[d], bv = bbS[d];
            // X: fragment rows are consecutive q -> one 16B store per lane.
            v8bf xv;
#pragma unroll
            for (int j = 0; j < 8; j++) xv[j] = f2bf(accx[j] * sx + hx);
            *(v8bf*)(X + d * M_ + n * Q_ + q0 + 8 * hi) = xv;
            // a/b: stage into LDS [q_local][d] for t-grouped coalesced stores.
            int ql = tl * 16 + 8 * hi;
#pragma unroll
            for (int j = 0; j < 8; j++) {
                LA[(ql + j) * 65 + d] = f2bf(acca[j] + av);
                LB[(ql + j) * 65 + d] = f2bf(accb[j] + bv);
            }
        }
    }
    __syncthreads();

    // Cooperative store: for each (d,v) write 16 consecutive nt (32 B runs).
    for (int p = tid; p < 1600; p += 256) {
        int d = p / 25, v = p - d * 25;
        v8bf lo = {}, hi8 = {};
#pragma unroll
        for (int t = 0; t < 8; t++) lo[t]  = LA[(t * 25 + v) * 65 + d];
#pragma unroll
        for (int t = 0; t < 8; t++) hi8[t] = LA[((t + 8) * 25 + v) * 65 + d];
        __bf16* dst = A1 + (d * V_ + v) * NT_ + ntb;
        *(v8bf*)dst = lo;  *(v8bf*)(dst + 8) = hi8;
#pragma unroll
        for (int t = 0; t < 8; t++) lo[t]  = LB[(t * 25 + v) * 65 + d];
#pragma unroll
        for (int t = 0; t < 8; t++) hi8[t] = LB[((t + 8) * 25 + v) * 65 + d];
        dst = B1 + (d * V_ + v) * NT_ + ntb;
        *(v8bf*)dst = lo;  *(v8bf*)(dst + 8) = hi8;
    }
}

// ---------------------------------------------------------------------------
__global__ __launch_bounds__(256) void k_zero(float* __restrict__ p, int n) {
    int i = blockIdx.x * 256 + threadIdx.x;
    if (i < n) p[i] = 0.f;
}

// ---------------------------------------------------------------------------
// Kernel 2: ATp[d,v,w] += partial( sum_nt A1[d,v,nt] * B1[d,w,nt] )
// 512 blocks = 64 d x 8 K-slices; 8 waves split each slice; LDS reduction,
// then f32 atomics into the pre-tanh accumulator.
// ---------------------------------------------------------------------------
__global__ __launch_bounds__(256) void k_attn(
    const __bf16* __restrict__ A1, const __bf16* __restrict__ B1,
    float* __restrict__ ATp)
{
    __shared__ float red[8][4][8][32];   // 32 KB
    int d = blockIdx.x >> 3, slice = blockIdx.x & 7;
    int tid = threadIdx.x, wave = tid >> 5, lane = tid & 31;
    int r = lane & 15, hi = lane >> 4;

    v8f acc[4] = {};
    int kbeg = (slice * 8 + wave) * 256;
    for (int k0 = kbeg; k0 < kbeg + 256; k0 += 32) {
        v16bf afr[2], bfr[2];
#pragma unroll
        for (int t2 = 0; t2 < 2; t2++) {
            int vv = t2 * 16 + r;
            v16bf fa = {}, fb = {};
            if (vv < V_) {
                const __bf16* pa = A1 + (d * V_ + vv) * NT_ + k0 + hi * 8;
                const __bf16* pb = B1 + (d * V_ + vv) * NT_ + k0 + hi * 8;
                v8bf a0 = *(const v8bf*)pa, a1 = *(const v8bf*)(pa + 16);
                v8bf b0 = *(const v8bf*)pb, b1 = *(const v8bf*)(pb + 16);
                fa = __builtin_shufflevector(a0, a1, 0,1,2,3,4,5,6,7,8,9,10,11,12,13,14,15);
                fb = __builtin_shufflevector(b0, b1, 0,1,2,3,4,5,6,7,8,9,10,11,12,13,14,15);
            }
            afr[t2] = fa; bfr[t2] = fb;
        }
        acc[0] = wmma_bf16(afr[0], bfr[0], acc[0]);
        acc[1] = wmma_bf16(afr[0], bfr[1], acc[1]);
        acc[2] = wmma_bf16(afr[1], bfr[0], acc[2]);
        acc[3] = wmma_bf16(afr[1], bfr[1], acc[3]);
    }
#pragma unroll
    for (int t2 = 0; t2 < 4; t2++)
#pragma unroll
        for (int j = 0; j < 8; j++) red[wave][t2][j][lane] = acc[t2][j];
    __syncthreads();

    for (int idx = tid; idx < 1024; idx += 256) {
        int t2 = idx >> 8, j = (idx >> 5) & 7, ln = idx & 31;
        float s = 0.f;
#pragma unroll
        for (int w8 = 0; w8 < 8; w8++) s += red[w8][t2][j][ln];
        int v = (t2 >> 1) * 16 + j + 8 * (ln >> 4);
        int w = (t2 & 1) * 16 + (ln & 15);
        if (v < V_ && w < V_)
            atomicAdd(&ATp[(d * V_ + v) * V_ + w], s);
    }
}

// ---------------------------------------------------------------------------
// Kernel 3: AM[d][w][v] = sum_c wc[d,c]*tanh(ATp[c,v,w]/V) + bc[d]
// bf16, zero-padded to 32x32, column-major => ready-made B fragments.
// ---------------------------------------------------------------------------
__global__ __launch_bounds__(256) void k_mix(
    const float* __restrict__ ATp, const float* __restrict__ wc,
    const float* __restrict__ bc, __bf16* __restrict__ AM)
{
    int idx = blockIdx.x * 256 + threadIdx.x;   // 64*32*32 = 65536
    int v = idx & 31, w = (idx >> 5) & 31, d = idx >> 10;
    float s = 0.f;
    if (v < V_ && w < V_) {
        s = bc[d];
        for (int c = 0; c < 64; c++)
            s += wc[d * 64 + c] * tanhf(ATp[(c * V_ + v) * V_ + w] * (1.0f / V_));
    }
    AM[(d * 32 + w) * 32 + v] = f2bf(s);
}

// ---------------------------------------------------------------------------
// Kernel 4: out[n,d,t,w] = relu( bn( sum_v X[d,nt,v]*AM[d,v,w] ) + x0 )
// Per-channel 2 KB adjacency staged via async copy to LDS; single K=32 WMMA.
// ---------------------------------------------------------------------------
__global__ __launch_bounds__(256) void k_apply(
    const __bf16* __restrict__ X, const __bf16* __restrict__ AM,
    const float* __restrict__ x0,
    const float* __restrict__ bng, const float* __restrict__ bnb,
    const float* __restrict__ bnm, const float* __restrict__ bnv,
    float* __restrict__ out)
{
    __shared__ __bf16 am[1024];   // AM[d] 32x32 = 2 KB
    __shared__ float  ss[2];
    int tid = threadIdx.x;
    int d = blockIdx.x >> 7;      // 128 blocks per channel

    // Async byte-copy AM[d] -> LDS (CDNA5 async path, tracked by ASYNCcnt).
    if (tid < 128) {
        unsigned lds = (unsigned)(unsigned long long)(&am[0]) + (unsigned)tid * 16u;
        const __bf16* g = AM + d * 1024 + tid * 8;
        asm volatile("global_load_async_to_lds_b128 %0, %1, off"
                     :: "v"(lds), "v"(g) : "memory");
    }
    if (tid == 0) {
        float inv = bng[d] * rsqrtf(bnv[d] + 1e-5f);
        ss[0] = inv; ss[1] = bnb[d] - bnm[d] * inv;
    }
    asm volatile("s_wait_asynccnt 0x0" ::: "memory");
    __syncthreads();

    int wave = tid >> 5, lane = tid & 31, r = lane & 15, hi = lane >> 4;
    int nt0 = ((blockIdx.x & 127) * 8 + wave) * 16;

    // A fragment: 16 rows of X[d] (25 bf16 each), K = v padded to 32.
    v16bf af = {};
    const __bf16* xrow = X + d * M_ + (nt0 + r) * V_;
#pragma unroll
    for (int e = 0; e < 16; e++) {
        int v = ((e & 8) << 1) + hi * 8 + (e & 7);
        if (v < V_) af[e] = xrow[v];
    }

    v8f acc[2];
#pragma unroll
    for (int wt = 0; wt < 2; wt++) {
        v16bf bf_ = ldb64(am + (wt * 16 + r) * 32, 0, hi) ;
        // note: am rows are stride 32, ldb64 assumes stride 64 -> inline here:
        v16bf bb_;
        const __bf16* arow = am + (wt * 16 + r) * 32;
#pragma unroll
        for (int e = 0; e < 8; e++) { bb_[e] = arow[hi * 8 + e]; bb_[8 + e] = arow[16 + hi * 8 + e]; }
        (void)bf_;
        v8f z = {};
        acc[wt] = wmma_bf16(af, bb_, z);
    }

    float scl = ss[0], sh = ss[1];
#pragma unroll
    for (int wt = 0; wt < 2; wt++) {
        int w = wt * 16 + r;
#pragma unroll
        for (int j = 0; j < 8; j++) {
            int ntr = nt0 + j + 8 * hi;
            if (w < V_) {
                int idx = ((ntr >> 8) * D_ + d) * Q_ + (ntr & 255) * V_ + w;
                float val = acc[wt][j] * scl + sh + x0[idx];
                out[idx] = fmaxf(val, 0.f);
            }
        }
    }
}

// ---------------------------------------------------------------------------
extern "C" void kernel_launch(void* const* d_in, const int* in_sizes, int n_in,
                              void* d_out, int out_size, void* d_ws, size_t ws_size,
                              hipStream_t stream) {
    const float* x0    = (const float*)d_in[0];
    const float* lin_w = (const float*)d_in[1];
    const float* lin_b = (const float*)d_in[2];
    const float* wa    = (const float*)d_in[3];
    const float* ba    = (const float*)d_in[4];
    const float* wb    = (const float*)d_in[5];
    const float* bb    = (const float*)d_in[6];
    const float* wc    = (const float*)d_in[7];
    const float* bc    = (const float*)d_in[8];
    const float* bn0g  = (const float*)d_in[9];
    const float* bn0b  = (const float*)d_in[10];
    const float* bn0m  = (const float*)d_in[11];
    const float* bn0v  = (const float*)d_in[12];
    const float* bng   = (const float*)d_in[13];
    const float* bnb   = (const float*)d_in[14];
    const float* bnm   = (const float*)d_in[15];
    const float* bnv   = (const float*)d_in[16];
    float* outp = (float*)d_out;

    // workspace: X | A1 | B1 (bf16, 52,428,800 B each) | ATp f32 | AM bf16
    char* ws = (char*)d_ws;
    const size_t SZ = (size_t)D_ * M_ * 2;
    __bf16* X   = (__bf16*)(ws);
    __bf16* A1  = (__bf16*)(ws + SZ);
    __bf16* B1  = (__bf16*)(ws + 2 * SZ);
    float*  ATp = (float*) (ws + 3 * SZ);
    __bf16* AM  = (__bf16*)(ws + 3 * SZ + 160000);

    k_gemm3<<<1024, 256, 129600, stream>>>(x0, lin_w, lin_b, wa, ba, wb, bb,
                                           bn0g, bn0b, bn0m, bn0v, X, A1, B1);
    k_zero <<<157,  256, 0, stream>>>(ATp, 40000);
    k_attn <<<512,  256, 0, stream>>>(A1, B1, ATp);
    k_mix  <<<256,  256, 0, stream>>>(ATp, wc, bc, AM);
    k_apply<<<8192, 256, 0, stream>>>(X, AM, x0, bng, bnb, bnm, bnv, outp);
}
